// TwoLayer_HierarchicalSNN_36292473651223
// MI455X (gfx1250) — matrix operand order repaired and checked by reading, not compiled, not run on hardware
//
#include <hip/hip_runtime.h>
#include <hip/hip_bf16.h>

namespace {

constexpr int kB   = 256;
constexpr int kT   = 128;
constexpr int kIN  = 1024;
constexpr int kH0  = 2048;
constexpr int kH1  = 2048;
constexpr int kOUT = 512;
constexpr float kBETA = 0.9f;
constexpr float kTHR  = 1.0f;

typedef __attribute__((ext_vector_type(8)))  float          v8f;
typedef __attribute__((ext_vector_type(4)))  float          v4f;
typedef __attribute__((ext_vector_type(16))) __bf16         v16bf;
typedef __attribute__((ext_vector_type(16))) unsigned short v16u;
typedef __attribute__((ext_vector_type(8)))  unsigned short v8u;

__device__ __forceinline__ unsigned short f2bf(float f) {
  unsigned int u = __float_as_uint(f);
  u += 0x7fffu + ((u >> 16) & 1u);   // round-to-nearest-even
  return (unsigned short)(u >> 16);
}
__device__ __forceinline__ float bf2f(unsigned short h) {
  return __uint_as_float((unsigned int)h << 16);
}

__device__ __forceinline__ v8f wmma_bf16(v16u a, v16u b, v8f c) {
  return __builtin_amdgcn_wmma_f32_16x16x32_bf16(
      false, __builtin_bit_cast(v16bf, a),
      false, __builtin_bit_cast(v16bf, b),
      (short)0, c, false, false);
}

// A operand (16x32 bf16, M x K): lane L(<16): row m0+L, K = {k..k+7, k+16..k+23}
//                                lane L(>=16): row m0+L-16, K = {k+8..k+15, k+24..k+31}
__device__ __forceinline__ v16u load_a_bf16(const unsigned short* __restrict__ row,
                                            int k, int half) {
  const v8u lo = *(const v8u*)(row + k + 8 * half);
  const v8u hi = *(const v8u*)(row + k + 16 + 8 * half);
  v16u r;
#pragma unroll
  for (int i = 0; i < 8; ++i) { r[i] = lo[i]; r[8 + i] = hi[i]; }
  return r;
}

// B operand (32x16 bf16, K x N) with B[k][n] = W[n][k] (row-major W[N][K]):
// lane L(<16): col n0+L, K = {k..k+7, k+16..k+23} ; lane L(>=16): col n0+L-16, K = {k+8..k+15, k+24..k+31}
__device__ __forceinline__ v16u load_b_bf16(const unsigned short* __restrict__ wrow,
                                            int k, int half) {
  const unsigned short* p = wrow + k + 8 * half;
  const v8u lo = *(const v8u*)(p);
  const v8u hi = *(const v8u*)(p + 16);
  v16u r;
#pragma unroll
  for (int i = 0; i < 8; ++i) { r[i] = lo[i]; r[8 + i] = hi[i]; }
  return r;
}

// 2x2 register-blocked GEMM: wave computes a 32x32 tile of Act[M][K] @ W[N][K]^T.
// 4 operand loads feed 4 WMMAs -> 16 FLOP/byte from L2 (before WGP$ sharing).
__device__ __forceinline__ void gemm22(const unsigned short* __restrict__ arow,
                                       const unsigned short* __restrict__ brow,
                                       int K, long strideA, long strideB,
                                       int half, v8f acc[2][2]) {
#pragma unroll 2
  for (int k = 0; k < K; k += 32) {
    const v16u a0 = load_a_bf16(arow, k, half);
    const v16u a1 = load_a_bf16(arow + 16 * strideA, k, half);
    const v16u b0 = load_b_bf16(brow, k, half);
    const v16u b1 = load_b_bf16(brow + 16 * strideB, k, half);
    acc[0][0] = wmma_bf16(a0, b0, acc[0][0]);
    acc[0][1] = wmma_bf16(a0, b1, acc[0][1]);
    acc[1][0] = wmma_bf16(a1, b0, acc[1][0]);
    acc[1][1] = wmma_bf16(a1, b1, acc[1][1]);
  }
}

// Block = 8 waves arranged 4(m) x 2(n): macro-tile 128x64. Waves sharing an
// m-row hit the same A tile in WGP$; waves sharing an n-col share B rows.
// Balances L2 traffic: A*(N/64) + B*(M/128) instead of B-dominated streaming.
__device__ __forceinline__ void tile_coords(int nTiles64, int& m0, int& n0) {
  const int wib = (int)(threadIdx.x >> 5);         // 0..7
  const int bm  = (int)blockIdx.x / nTiles64;      // m-block
  const int bn  = (int)blockIdx.x % nTiles64;      // n-block
  m0 = (bm * 4 + (wib >> 1)) * 32;
  n0 = (bn * 2 + (wib & 1)) * 32;
}

// Vectorized fp32 -> bf16 (with scale); n must be a multiple of 8.
__global__ __launch_bounds__(256) void k_cvt8(const float* __restrict__ s,
                                              unsigned short* __restrict__ d,
                                              int n8, float scale) {
  const int i = blockIdx.x * blockDim.x + threadIdx.x;
  if (i < n8) {
    const v4f a = ((const v4f*)s)[2 * i];
    const v4f b = ((const v4f*)s)[2 * i + 1];
    v8u r;
#pragma unroll
    for (int j = 0; j < 4; ++j) {
      r[j]     = f2bf(a[j] * scale);
      r[4 + j] = f2bf(b[j] * scale);
    }
    ((v8u*)d)[i] = r;
  }
}

__global__ __launch_bounds__(256) void k_zero(unsigned int* __restrict__ p, int n) {
  int i = blockIdx.x * blockDim.x + threadIdx.x;
  if (i < n) p[i] = 0u;
}

// ---- Layer 0: v0n = (1-b0)*(BETA*v0 + z0@Wzv0^T + bzv0) + (x_t@Wbv0^T + bbv0)
//               + b1*(z1@Wtv0^T + btv0) - z0*THR ; z0n = (v0n-THR)>0
__global__ __launch_bounds__(256) void k_l0(
    const unsigned short* __restrict__ xbf, int t,   // [B][T][IN] bf16
    const unsigned short* __restrict__ Wbv0,         // [H0][IN]  bf16
    const unsigned short* __restrict__ Wzv0,         // [H0][H0]  bf16
    const unsigned short* __restrict__ Wtv0,         // [H0][H1]  bf16
    const float* __restrict__ bbv0, const float* __restrict__ bzv0,
    const float* __restrict__ btv0,
    const unsigned short* __restrict__ z0in,         // [B][H0] bf16 spikes
    const unsigned short* __restrict__ z1in,         // [B][H1] bf16 spikes
    const float* __restrict__ b0v, const float* __restrict__ b1v,
    float* __restrict__ v0,                          // [B][H0] in-place
    unsigned short* __restrict__ z0out) {
  const int lane = (int)(threadIdx.x & 31);
  const int half = lane >> 4, r15 = lane & 15;
  int m0, n0;
  tile_coords(kH0 / 64, m0, n0);

  const unsigned short* xrow  = xbf  + (size_t)(m0 + r15) * (kT * kIN) + (size_t)t * kIN;
  const unsigned short* z0row = z0in + (size_t)(m0 + r15) * kH0;
  const unsigned short* z1row = z1in + (size_t)(m0 + r15) * kH1;
  const unsigned short* wb    = Wbv0 + (size_t)(n0 + r15) * kIN;
  const unsigned short* wz    = Wzv0 + (size_t)(n0 + r15) * kH0;
  const unsigned short* wt    = Wtv0 + (size_t)(n0 + r15) * kH1;

  v8f ax[2][2] = {}, az[2][2] = {}, at[2][2] = {};
  gemm22(xrow,  wb, kIN, (long)kT * kIN, kIN, half, ax);
  gemm22(z0row, wz, kH0, kH0,            kH0, half, az);
  gemm22(z1row, wt, kH1, kH1,            kH1, half, at);

#pragma unroll
  for (int tn = 0; tn < 2; ++tn) {
    const int n = n0 + 16 * tn + r15;
    const float bb = bbv0[n], bz = bzv0[n], bt = btv0[n];
#pragma unroll
    for (int tm = 0; tm < 2; ++tm) {
#pragma unroll
      for (int r = 0; r < 8; ++r) {
        const int m = m0 + 16 * tm + r + 8 * half;   // C/D layout: VGPR r, lane halves
        const size_t idx = (size_t)m * kH0 + n;
        const float vold = v0[idx];
        const float zold = bf2f(z0in[idx]);
        const float b0m = b0v[m], b1m = b1v[m];
        const float vn = (1.0f - b0m) * (kBETA * vold + (az[tm][tn][r] + bz))
                       + (ax[tm][tn][r] + bb) + b1m * (at[tm][tn][r] + bt) - zold * kTHR;
        v0[idx] = vn;
        z0out[idx] = (vn - kTHR) > 0.0f ? (unsigned short)0x3F80u : (unsigned short)0u;
      }
    }
  }
}

// ---- Layer 1: v1n = BETA*v1 + (z1@Wzv1^T+bzv1) + b0n*(z0n@Wbv1^T+bbv1) - z1*THR
__global__ __launch_bounds__(256) void k_l1(
    const unsigned short* __restrict__ Wzv1,   // [H1][H1] bf16
    const unsigned short* __restrict__ Wbv1,   // [H1][H0] bf16
    const float* __restrict__ bzv1, const float* __restrict__ bbv1,
    const unsigned short* __restrict__ z1in,
    const unsigned short* __restrict__ z0new,
    const float* __restrict__ b0v,
    float* __restrict__ v1,
    unsigned short* __restrict__ z1out,
    float* __restrict__ zsum) {
  const int lane = (int)(threadIdx.x & 31);
  const int half = lane >> 4, r15 = lane & 15;
  int m0, n0;
  tile_coords(kH1 / 64, m0, n0);

  const unsigned short* z1row = z1in  + (size_t)(m0 + r15) * kH1;
  const unsigned short* z0row = z0new + (size_t)(m0 + r15) * kH0;
  const unsigned short* wz    = Wzv1  + (size_t)(n0 + r15) * kH1;
  const unsigned short* wb    = Wbv1  + (size_t)(n0 + r15) * kH0;

  v8f az[2][2] = {}, ab[2][2] = {};
  gemm22(z1row, wz, kH1, kH1, kH1, half, az);
  gemm22(z0row, wb, kH0, kH0, kH0, half, ab);

#pragma unroll
  for (int tn = 0; tn < 2; ++tn) {
    const int n = n0 + 16 * tn + r15;
    const float bz = bzv1[n], bb = bbv1[n];
#pragma unroll
    for (int tm = 0; tm < 2; ++tm) {
#pragma unroll
      for (int r = 0; r < 8; ++r) {
        const int m = m0 + 16 * tm + r + 8 * half;
        const size_t idx = (size_t)m * kH1 + n;
        const float vold = v1[idx];
        const float zold = bf2f(z1in[idx]);
        const float b0m = b0v[m];
        const float vn = kBETA * vold + (az[tm][tn][r] + bz)
                       + b0m * (ab[tm][tn][r] + bb) - zold * kTHR;
        v1[idx] = vn;
        const float zn = (vn - kTHR) > 0.0f ? 1.0f : 0.0f;
        z1out[idx] = zn > 0.0f ? (unsigned short)0x3F80u : (unsigned short)0u;
        zsum[idx] += zn;
      }
    }
  }
}

// ---- branch gate: b = (dot(z[m,:], w) + bias > 0)   [sigmoid(s)-0.5>0 <=> s>0]
__global__ __launch_bounds__(256) void k_branch(
    const unsigned short* __restrict__ z, const float* __restrict__ w,
    const float* __restrict__ bias, float* __restrict__ out, int H) {
  __shared__ float red[256];
  const int m = blockIdx.x;
  float acc = 0.0f;
  for (int n = threadIdx.x; n < H; n += 256)
    acc += bf2f(z[(size_t)m * H + n]) * w[n];
  red[threadIdx.x] = acc;
  __syncthreads();
  for (int off = 128; off > 0; off >>= 1) {
    if ((int)threadIdx.x < off) red[threadIdx.x] += red[threadIdx.x + off];
    __syncthreads();
  }
  if (threadIdx.x == 0) out[m] = (red[0] + bias[0]) > 0.0f ? 1.0f : 0.0f;
}

// ---- final FC: out = zs_bf @ Wfc^T + bfc   (zs_bf = bf16(zsum/T))
__global__ __launch_bounds__(256) void k_fc(
    const unsigned short* __restrict__ zsbf,  // [B][H1] bf16 (pre-scaled by 1/T)
    const unsigned short* __restrict__ Wfc,   // [OUT][H1] bf16
    const float* __restrict__ bfc,
    float* __restrict__ out) {
  const int lane = (int)(threadIdx.x & 31);
  const int half = lane >> 4, r15 = lane & 15;
  int m0, n0;
  tile_coords(kOUT / 64, m0, n0);

  const unsigned short* arow = zsbf + (size_t)(m0 + r15) * kH1;
  const unsigned short* wrow = Wfc  + (size_t)(n0 + r15) * kH1;

  v8f acc[2][2] = {};
  gemm22(arow, wrow, kH1, kH1, kH1, half, acc);

#pragma unroll
  for (int tn = 0; tn < 2; ++tn) {
    const int n = n0 + 16 * tn + r15;
    const float b = bfc[n];
#pragma unroll
    for (int tm = 0; tm < 2; ++tm) {
#pragma unroll
      for (int r = 0; r < 8; ++r) {
        const int m = m0 + 16 * tm + r + 8 * half;
        out[(size_t)m * kOUT + n] = acc[tm][tn][r] + b;
      }
    }
  }
}

} // anonymous namespace

extern "C" void kernel_launch(void* const* d_in, const int* in_sizes, int n_in,
                              void* d_out, int out_size, void* d_ws, size_t ws_size,
                              hipStream_t stream) {
  (void)in_sizes; (void)n_in; (void)out_size; (void)ws_size;
  const float* x     = (const float*)d_in[0];
  const float* Wbv0f = (const float*)d_in[1];
  const float* bbv0  = (const float*)d_in[2];
  const float* Wtv0f = (const float*)d_in[3];
  const float* btv0  = (const float*)d_in[4];
  const float* Wzv0f = (const float*)d_in[5];
  const float* bzv0  = (const float*)d_in[6];
  const float* Wzb0  = (const float*)d_in[7];
  const float* bzb0  = (const float*)d_in[8];
  const float* Wbv1f = (const float*)d_in[9];
  const float* bbv1  = (const float*)d_in[10];
  const float* Wzv1f = (const float*)d_in[11];
  const float* bzv1  = (const float*)d_in[12];
  const float* Wzb1  = (const float*)d_in[13];
  const float* bzb1  = (const float*)d_in[14];
  const float* Wfcf  = (const float*)d_in[15];
  const float* bfc   = (const float*)d_in[16];
  float* out = (float*)d_out;

  char* ws = (char*)d_ws;
  size_t off = 0;
  auto take = [&](size_t bytes) -> char* {
    char* p = ws + off;
    off = (off + bytes + 255) & ~(size_t)255;
    return p;
  };
  unsigned short* Wbv0 = (unsigned short*)take((size_t)kH0 * kIN * 2);
  unsigned short* Wzv0 = (unsigned short*)take((size_t)kH0 * kH0 * 2);
  unsigned short* Wtv0 = (unsigned short*)take((size_t)kH0 * kH1 * 2);
  unsigned short* Wbv1 = (unsigned short*)take((size_t)kH1 * kH0 * 2);
  unsigned short* Wzv1 = (unsigned short*)take((size_t)kH1 * kH1 * 2);
  unsigned short* Wfc  = (unsigned short*)take((size_t)kOUT * kH1 * 2);
  unsigned short* xbf  = (unsigned short*)take((size_t)kB * kT * kIN * 2);
  unsigned short* zsbf = (unsigned short*)take((size_t)kB * kH1 * 2);
  char* state0 = ws + off;
  float*          v0   = (float*)take((size_t)kB * kH0 * 4);
  float*          v1   = (float*)take((size_t)kB * kH1 * 4);
  unsigned short* z0a  = (unsigned short*)take((size_t)kB * kH0 * 2);
  unsigned short* z0b  = (unsigned short*)take((size_t)kB * kH0 * 2);
  unsigned short* z1a  = (unsigned short*)take((size_t)kB * kH1 * 2);
  unsigned short* z1b  = (unsigned short*)take((size_t)kB * kH1 * 2);
  float*          b0   = (float*)take((size_t)kB * 4);
  float*          b1   = (float*)take((size_t)kB * 4);
  float*          zsum = (float*)take((size_t)kB * kH1 * 4);
  const size_t stateBytes = (size_t)((ws + off) - state0);

  const dim3 blk(256);
  auto cvt = [&](const float* s, unsigned short* d, size_t n, float scale) {
    const size_t n8 = n / 8;
    k_cvt8<<<dim3((unsigned)((n8 + 255) / 256)), blk, 0, stream>>>(s, d, (int)n8, scale);
  };
  cvt(Wbv0f, Wbv0, (size_t)kH0 * kIN, 1.0f);
  cvt(Wzv0f, Wzv0, (size_t)kH0 * kH0, 1.0f);
  cvt(Wtv0f, Wtv0, (size_t)kH0 * kH1, 1.0f);
  cvt(Wbv1f, Wbv1, (size_t)kH1 * kH0, 1.0f);
  cvt(Wzv1f, Wzv1, (size_t)kH1 * kH1, 1.0f);
  cvt(Wfcf,  Wfc,  (size_t)kOUT * kH1, 1.0f);
  cvt(x,     xbf,  (size_t)kB * kT * kIN, 1.0f);

  k_zero<<<dim3((unsigned)((stateBytes / 4 + 255) / 256)), blk, 0, stream>>>(
      (unsigned int*)state0, (int)(stateBytes / 4));

  unsigned short *z0c = z0a, *z0n = z0b, *z1c = z1a, *z1n = z1b;
  // blocks: (M/128) * (N/64) ; 8 waves (4m x 2n) per block
  const dim3 gemmGrid((kB / 128) * (kH0 / 64));      // 2*32 = 64 blocks
  for (int t = 0; t < kT; ++t) {
    k_l0<<<gemmGrid, blk, 0, stream>>>(xbf, t, Wbv0, Wzv0, Wtv0, bbv0, bzv0, btv0,
                                       z0c, z1c, b0, b1, v0, z0n);
    k_branch<<<dim3(kB), blk, 0, stream>>>(z0n, Wzb0, bzb0, b0, kH0);
    k_l1<<<gemmGrid, blk, 0, stream>>>(Wzv1, Wbv1, bzv1, bbv1, z1c, z0n, b0,
                                       v1, z1n, zsum);
    k_branch<<<dim3(kB), blk, 0, stream>>>(z1n, Wzb1, bzb1, b1, kH1);
    unsigned short* tmp;
    tmp = z0c; z0c = z0n; z0n = tmp;
    tmp = z1c; z1c = z1n; z1n = tmp;
  }
  cvt(zsum, zsbf, (size_t)kB * kH1, 1.0f / (float)kT);
  k_fc<<<dim3((kB / 128) * (kOUT / 64)), blk, 0, stream>>>(zsbf, Wfc, bfc, out);
}
